// GraphLevelReadout_67250597921411
// MI455X (gfx1250) — matrix operand order repaired and checked by reading, not compiled, not run on hardware
//
#include <hip/hip_runtime.h>

// ---------------------------------------------------------------------------
// Types for CDNA5 WMMA (wave32): v_wmma_f32_16x16x32_bf16
// ---------------------------------------------------------------------------
typedef __attribute__((ext_vector_type(4)))  __bf16 v4bf;
typedef __attribute__((ext_vector_type(8)))  __bf16 v8bf;
typedef __attribute__((ext_vector_type(16))) __bf16 v16bf;
typedef __attribute__((ext_vector_type(8)))  float  v8f;

#define IN_F   256
#define HID    512
#define OUT_F  256
#define NGRAPH 2048
#define NNODES 200000
#define TMN    64     // node rows per block (node kernel)
#define TMG    32     // graph rows per block (graph kernel)

// LDS pitches (elements), padded so row stride ≡ 4 banks -> conflict-free frag loads
#define PX   264   // bf16 pitch for 256-wide tile (528 B/row)
#define PH   520   // bf16 pitch for 512-wide tile (1040 B/row)

// ---------------------------------------------------------------------------
// Fragment loaders per CDNA5 ISA 7.12.2 layouts.
// A (16-bit, 16x32):  lanes 0-15 row M=lane hold K {k0..k0+7} U {k0+16..k0+23}
//                     lanes16-31 row M=lane-16 hold K {k0+8..k0+15} U {k0+24..k0+31}
// B (16-bit, 32x16):  lanes 0-15 col N=lane hold K {k0..k0+7} U {k0+16..k0+23}
//                     lanes16-31 col N=lane-16 hold K {k0+8..k0+15} U {k0+24..k0+31}
// (B is loaded from a pre-transposed weight matrix W^T stored [N, K] row-major.)
// ---------------------------------------------------------------------------
__device__ __forceinline__ v16bf load_fragA(const __bf16* base, int pitch,
                                            int rbase, int k0, int lane) {
    const int r   = rbase + (lane & 15);
    const int kof = k0 + ((lane >> 4) << 3);
    const __bf16* p = base + r * pitch + kof;
    v8bf lo = *(const v8bf*)(p);
    v8bf hi = *(const v8bf*)(p + 16);
    return __builtin_shufflevector(lo, hi, 0,1,2,3,4,5,6,7,8,9,10,11,12,13,14,15);
}

__device__ __forceinline__ v16bf load_fragB(const __bf16* base, int pitch,
                                            int nbase, int k0, int lane) {
    const int n   = nbase + (lane & 15);
    const int kof = k0 + ((lane >> 4) << 3);
    const __bf16* p = base + n * pitch + kof;
    v8bf lo = *(const v8bf*)(p);
    v8bf hi = *(const v8bf*)(p + 16);
    return __builtin_shufflevector(lo, hi, 0,1,2,3,4,5,6,7,8,9,10,11,12,13,14,15);
}

__device__ __forceinline__ v8f wmma_bf16(v16bf a, v16bf b, v8f c) {
    return __builtin_amdgcn_wmma_f32_16x16x32_bf16(false, a, false, b,
                                                   (short)0, c, false, false);
}

// ---------------------------------------------------------------------------
// Prep kernels
// ---------------------------------------------------------------------------
__global__ void zero_f32_kernel(float* __restrict__ p, int n) {
    int i = blockIdx.x * 256 + threadIdx.x;
    if (i < n) p[i] = 0.0f;
}

// W [K,N] f32 row-major  ->  WT [N,K] bf16 row-major
__global__ void transpose_bf16_kernel(const float* __restrict__ W,
                                      __bf16* __restrict__ WT, int K, int N) {
    int idx = blockIdx.x * 256 + threadIdx.x;
    if (idx < K * N) {
        int k = idx / N;
        int n = idx - k * N;
        WT[n * K + k] = (__bf16)W[idx];
    }
}

// ---------------------------------------------------------------------------
// Node kernel: h = relu(relu(x@W1+b1)@W2+b2); segment-sum into pooled.
// 256 threads = 8 waves. Each wave owns a full-height 64-column stripe:
// 4 row-tiles x 4 n-tiles of 16x16 WMMA accumulators. Every B fragment is
// loaded once per block and reused across all 4 row-tiles.
// GEMM2 output is written back into sH1 (bf16) after a barrier, keeping LDS
// at ~100 KiB so TWO workgroups co-reside per WGP (320 KiB) and the staging /
// epilogue / reduction phases of one overlap the WMMA stream of the other.
// ---------------------------------------------------------------------------
__global__ __launch_bounds__(256) void node_mlp_kernel(
    const float* __restrict__ x, const int* __restrict__ seg,
    const __bf16* __restrict__ W1T, const float* __restrict__ b1,
    const __bf16* __restrict__ W2T, const float* __restrict__ b2,
    float* __restrict__ pooled)
{
    __shared__ __bf16 sX[TMN][PX];     // 33.0 KiB
    __shared__ __bf16 sH1[TMN][PH];    // 65.0 KiB  (holds H1, then reused for H2)
    __shared__ int    sSeg[TMN];

    const int tid  = threadIdx.x;
    const int lane = tid & 31;
    const int wid  = tid >> 5;
    const int nodeBase = blockIdx.x * TMN;

    // Stage x tile (64x256 f32) -> LDS bf16
    for (int i = tid; i < TMN * (IN_F / 4); i += 256) {
        int r  = i >> 6;          // IN_F/4 = 64 float4 per row
        int c4 = i & 63;
        float4 v = ((const float4*)x)[(nodeBase + r) * (IN_F / 4) + c4];
        v4bf b; b[0] = (__bf16)v.x; b[1] = (__bf16)v.y;
                b[2] = (__bf16)v.z; b[3] = (__bf16)v.w;
        *(v4bf*)(&sX[r][c4 * 4]) = b;
    }
    if (tid < TMN) sSeg[tid] = seg[nodeBase + tid];
    __syncthreads();

    const int cbase = wid * 64;       // wave's 64-column stripe
    const int nlo   = lane & 15;
    const int mhi   = (lane >> 4) << 3;

    // ---- GEMM1: (64x256) x (256x512) ----
    v8f acc[4][4];
    #pragma unroll
    for (int r = 0; r < 4; ++r)
        #pragma unroll
        for (int t = 0; t < 4; ++t) acc[r][t] = (v8f)0.0f;

    for (int k0 = 0; k0 < IN_F; k0 += 32) {
        v16bf a[4];
        #pragma unroll
        for (int r = 0; r < 4; ++r)
            a[r] = load_fragA(&sX[0][0], PX, r * 16, k0, lane);
        #pragma unroll
        for (int t = 0; t < 4; ++t) {
            v16bf b = load_fragB(W1T, IN_F, cbase + t * 16, k0, lane);
            #pragma unroll
            for (int r = 0; r < 4; ++r)
                acc[r][t] = wmma_bf16(a[r], b, acc[r][t]);
        }
    }
    #pragma unroll
    for (int t = 0; t < 4; ++t) {
        int col = cbase + t * 16 + nlo;
        float bias = b1[col];
        #pragma unroll
        for (int r = 0; r < 4; ++r)
            #pragma unroll
            for (int j = 0; j < 8; ++j) {
                float v = acc[r][t][j] + bias;
                v = v > 0.0f ? v : 0.0f;
                sH1[r * 16 + mhi + j][col] = (__bf16)v;
            }
    }
    __syncthreads();

    // ---- GEMM2: (64x512) x (512x512) ----
    v8f acc2[4][4];
    #pragma unroll
    for (int r = 0; r < 4; ++r)
        #pragma unroll
        for (int t = 0; t < 4; ++t) acc2[r][t] = (v8f)0.0f;

    for (int k0 = 0; k0 < HID; k0 += 32) {
        v16bf a[4];
        #pragma unroll
        for (int r = 0; r < 4; ++r)
            a[r] = load_fragA(&sH1[0][0], PH, r * 16, k0, lane);
        #pragma unroll
        for (int t = 0; t < 4; ++t) {
            v16bf b = load_fragB(W2T, HID, cbase + t * 16, k0, lane);
            #pragma unroll
            for (int r = 0; r < 4; ++r)
                acc2[r][t] = wmma_bf16(a[r], b, acc2[r][t]);
        }
    }

    // All waves finished reading sH1 -> safe to overwrite with H2 (bf16).
    __syncthreads();
    #pragma unroll
    for (int t = 0; t < 4; ++t) {
        int col = cbase + t * 16 + nlo;
        float bias = b2[col];
        #pragma unroll
        for (int r = 0; r < 4; ++r)
            #pragma unroll
            for (int j = 0; j < 8; ++j) {
                float v = acc2[r][t][j] + bias;
                v = v > 0.0f ? v : 0.0f;
                sH1[r * 16 + mhi + j][col] = (__bf16)v;
            }
    }
    __syncthreads();

    // ---- Segment-sum (ids sorted): run-length reduce down the 64 rows,
    //      f32 accumulation, one f32 atomic per (run, column).
    //      Thread owns cols tid and tid+256.
    #pragma unroll
    for (int h = 0; h < 2; ++h) {
        const int col = tid + h * 256;
        float run = 0.0f;
        int cur = sSeg[0];
        for (int i = 0; i < TMN; ++i) {
            int s = sSeg[i];
            if (s != cur) {
                atomicAdd(&pooled[cur * HID + col], run);
                run = 0.0f;
                cur = s;
            }
            run += (float)sH1[i][col];
        }
        atomicAdd(&pooled[cur * HID + col], run);
    }
}

// ---------------------------------------------------------------------------
// Graph kernel: out = relu(pooled@Wg1+bg1)@Wg2 + bg2   (2048 rows, 64 blocks)
// ---------------------------------------------------------------------------
__global__ __launch_bounds__(256) void graph_mlp_kernel(
    const float* __restrict__ pooled,
    const __bf16* __restrict__ Wg1T, const float* __restrict__ bg1,
    const __bf16* __restrict__ Wg2T, const float* __restrict__ bg2,
    float* __restrict__ out)
{
    __shared__ __bf16 sP[TMG][PH];
    __shared__ __bf16 sH[TMG][PH];

    const int tid  = threadIdx.x;
    const int lane = tid & 31;
    const int wid  = tid >> 5;
    const int gBase = blockIdx.x * TMG;

    // Stage pooled tile (32x512 f32) -> LDS bf16
    for (int i = tid; i < TMG * (HID / 4); i += 256) {
        int r  = i >> 7;          // HID/4 = 128 float4 per row
        int c4 = i & 127;
        float4 v = ((const float4*)pooled)[(gBase + r) * (HID / 4) + c4];
        v4bf b; b[0] = (__bf16)v.x; b[1] = (__bf16)v.y;
                b[2] = (__bf16)v.z; b[3] = (__bf16)v.w;
        *(v4bf*)(&sP[r][c4 * 4]) = b;
    }
    __syncthreads();

    const int nlo = lane & 15;
    const int mhi = (lane >> 4) << 3;

    // ---- GEMM1: (32x512) x (512x512), relu ----
    // Each wave owns a 64-col stripe over both row tiles (2 rt x 4 nt), B reused.
    {
        const int cbase = wid * 64;
        v8f acc[2][4];
        #pragma unroll
        for (int r = 0; r < 2; ++r)
            #pragma unroll
            for (int t = 0; t < 4; ++t) acc[r][t] = (v8f)0.0f;
        for (int k0 = 0; k0 < HID; k0 += 32) {
            v16bf a[2];
            #pragma unroll
            for (int r = 0; r < 2; ++r)
                a[r] = load_fragA(&sP[0][0], PH, r * 16, k0, lane);
            #pragma unroll
            for (int t = 0; t < 4; ++t) {
                v16bf b = load_fragB(Wg1T, HID, cbase + t * 16, k0, lane);
                #pragma unroll
                for (int r = 0; r < 2; ++r)
                    acc[r][t] = wmma_bf16(a[r], b, acc[r][t]);
            }
        }
        #pragma unroll
        for (int t = 0; t < 4; ++t) {
            int col = cbase + t * 16 + nlo;
            float bias = bg1[col];
            #pragma unroll
            for (int r = 0; r < 2; ++r)
                #pragma unroll
                for (int j = 0; j < 8; ++j) {
                    float v = acc[r][t][j] + bias;
                    v = v > 0.0f ? v : 0.0f;
                    sH[r * 16 + mhi + j][col] = (__bf16)v;
                }
        }
    }
    __syncthreads();

    // ---- GEMM2: (32x512) x (512x256), no relu, write f32 out ----
    // Wave owns a 32-col stripe over both row tiles (2 rt x 2 nt).
    {
        const int cbase = wid * 32;
        v8f acc[2][2];
        #pragma unroll
        for (int r = 0; r < 2; ++r)
            #pragma unroll
            for (int t = 0; t < 2; ++t) acc[r][t] = (v8f)0.0f;
        for (int k0 = 0; k0 < HID; k0 += 32) {
            v16bf a[2];
            #pragma unroll
            for (int r = 0; r < 2; ++r)
                a[r] = load_fragA(&sH[0][0], PH, r * 16, k0, lane);
            #pragma unroll
            for (int t = 0; t < 2; ++t) {
                v16bf b = load_fragB(Wg2T, HID, cbase + t * 16, k0, lane);
                #pragma unroll
                for (int r = 0; r < 2; ++r)
                    acc[r][t] = wmma_bf16(a[r], b, acc[r][t]);
            }
        }
        #pragma unroll
        for (int t = 0; t < 2; ++t) {
            int col = cbase + t * 16 + nlo;
            float bias = bg2[col];
            #pragma unroll
            for (int r = 0; r < 2; ++r)
                #pragma unroll
                for (int j = 0; j < 8; ++j) {
                    out[(gBase + r * 16 + mhi + j) * OUT_F + col] = acc[r][t][j] + bias;
                }
        }
    }
}

// ---------------------------------------------------------------------------
// Launch: zero pooled, transpose weights to bf16, node MLP + segsum, graph MLP
// ---------------------------------------------------------------------------
extern "C" void kernel_launch(void* const* d_in, const int* in_sizes, int n_in,
                              void* d_out, int out_size, void* d_ws, size_t ws_size,
                              hipStream_t stream) {
    const float* x    = (const float*)d_in[0];
    const int*   seg  = (const int*)d_in[1];
    const float* W1   = (const float*)d_in[2];
    const float* b1   = (const float*)d_in[3];
    const float* W2   = (const float*)d_in[4];
    const float* b2   = (const float*)d_in[5];
    const float* Wg1  = (const float*)d_in[6];
    const float* bg1  = (const float*)d_in[7];
    const float* Wg2  = (const float*)d_in[8];
    const float* bg2  = (const float*)d_in[9];
    float* out = (float*)d_out;

    char* ws = (char*)d_ws;
    float*  pooled = (float*)ws;                               // 2048*512*4 = 4 MiB
    __bf16* W1T  = (__bf16*)(ws + 4194304);                    // [512,256]
    __bf16* W2T  = (__bf16*)(ws + 4194304 + 262144);           // [512,512]
    __bf16* Wg1T = (__bf16*)(ws + 4194304 + 262144 + 524288);  // [512,512]
    __bf16* Wg2T = (__bf16*)(ws + 4194304 + 262144 + 2*524288);// [256,512]

    zero_f32_kernel<<<(NGRAPH * HID + 255) / 256, 256, 0, stream>>>(pooled, NGRAPH * HID);

    transpose_bf16_kernel<<<(IN_F * HID + 255) / 256, 256, 0, stream>>>(W1,  W1T,  IN_F, HID);
    transpose_bf16_kernel<<<(HID * HID + 255) / 256, 256, 0, stream>>>(W2,  W2T,  HID,  HID);
    transpose_bf16_kernel<<<(HID * HID + 255) / 256, 256, 0, stream>>>(Wg1, Wg1T, HID,  HID);
    transpose_bf16_kernel<<<(HID * OUT_F + 255) / 256, 256, 0, stream>>>(Wg2, Wg2T, HID, OUT_F);

    node_mlp_kernel<<<NNODES / TMN, 256, 0, stream>>>(x, seg, W1T, b1, W2T, b2, pooled);

    graph_mlp_kernel<<<NGRAPH / TMG, 256, 0, stream>>>(pooled, Wg1T, bg1, Wg2T, bg2, out);
}
